// Feature_Encoder_47029891891860
// MI455X (gfx1250) — matrix operand order repaired and checked by reading, compiled, not run on hardware
//
#include <hip/hip_runtime.h>
#include <hip/hip_bf16.h>

// ---------------------------------------------------------------------------
// Feature_Encoder for MI455X (gfx1250, wave32, WMMA + async LDS staging).
//   K0 city:    c[b][d] = static[b]·W_city[d] + b_city[d]
//   K1 conv1:   y1 = leaky(conv1d(history,w1,b1))        [32][128][2046] scalar
//   K2 featgemm x1 = leaky(input @ W_feat^T + b_feat)    [65152][512]   WMMA x8/wave
//   K3 conv2:   y2 = leaky(conv(maxpool3(y1),w2,b2))     [32][256][2042] WMMA,
//               X slab async-staged to LDS, pool fused on LDS read
//   K4 conv3:   y3 = leaky(conv(maxpool3(y2),w3,b3))     [32][512][2038] WMMA
//   K5 final:   out = LN2( LN1(x1 + maxpool3(y3)^T) * c )  block reductions
// ---------------------------------------------------------------------------

#define SEQ 2036
#define BB  32
#define NF  64
#define NC  100
#define DD  512
#define NO  4
#define LL  2048

typedef __attribute__((ext_vector_type(16))) _Float16 v16h;
typedef __attribute__((ext_vector_type(8)))  float    v8f;

__device__ __forceinline__ float leaky_f(float x) { return x > 0.0f ? x : 0.2f * x; }

// ------------------------------ K0: city embed ------------------------------
__global__ void city_kernel(const float* __restrict__ st, const float* __restrict__ Wc,
                            const float* __restrict__ bc, float* __restrict__ c) {
    int idx = blockIdx.x * blockDim.x + threadIdx.x;
    if (idx >= BB * DD) return;
    int d = idx & (DD - 1);
    int b = idx >> 9;
    float acc = bc[d];
    const float* s = st + b * NC;
    const float* w = Wc + d * NC;
    #pragma unroll 4
    for (int i = 0; i < NC; ++i) acc += s[i] * w[i];
    c[idx] = acc;
}

// ------------------------------ K1: conv1 (Cin=4, scalar) -------------------
__global__ void conv1_kernel(const float* __restrict__ hist, const float* __restrict__ w1,
                             const float* __restrict__ b1, float* __restrict__ y1) {
    // y1: [32][128][2046]
    const int L1 = LL - 2;
    size_t idx = (size_t)blockIdx.x * blockDim.x + threadIdx.x;
    const size_t total = (size_t)BB * 128 * L1;
    if (idx >= total) return;
    int t   = (int)(idx % L1);
    int tmp = (int)(idx / L1);
    int c1  = tmp & 127;
    int b   = tmp >> 7;
    const float* h = hist + (size_t)b * NO * LL;
    const float* w = w1 + c1 * (NO * 3);
    float acc = b1[c1];
    #pragma unroll
    for (int ci = 0; ci < NO; ++ci) {
        const float* hp = h + ci * LL + t;
        #pragma unroll
        for (int dk = 0; dk < 3; ++dk) acc += hp[dk] * w[ci * 3 + dk];
    }
    y1[idx] = leaky_f(acc);
}

// ------------------------------ K2: feature GEMM (WMMA f16) -----------------
// x1[m][d] = leaky(input @ W_feat^T + b_feat). Each wave: one 16-row m-tile,
// FOUR 16-col n-tiles (A fragment reused across 4 WMMAs per k-step).
__global__ void feat_gemm_kernel(const float* __restrict__ in, const float* __restrict__ Wf,
                                 const float* __restrict__ bf, float* __restrict__ x1) {
    const int lane   = threadIdx.x & 31;
    const int wave   = threadIdx.x >> 5;
    const int l15    = lane & 15;
    const int half16 = lane >> 4;
    const int m0  = (blockIdx.x * 4 + wave) * 16;  // 4072 m-tiles
    const int nb0 = blockIdx.y * 64;               // 8 n-groups of 4 tiles

    v8f acc0 = {}, acc1 = {}, acc2 = {}, acc3 = {};
    #pragma unroll
    for (int kb = 0; kb < NF; kb += 32) {
        // A fragment (16x32, contiguous k = 16*half16 + i; B below uses the same order)
        v16h a;
        {
            const float* ap = in + (size_t)(m0 + l15) * NF + kb + half16 * 16;
            #pragma unroll
            for (int i = 0; i < 8; ++i) a[i]     = (_Float16)ap[i];
            #pragma unroll
            for (int i = 0; i < 8; ++i) a[8 + i] = (_Float16)ap[8 + i];
        }
        #pragma unroll
        for (int tile = 0; tile < 4; ++tile) {
            v16h bfr;
            const float* bp = Wf + (size_t)(nb0 + tile * 16 + l15) * NF + kb + half16 * 16;
            __builtin_prefetch(bp + 64, 0, 1);
            #pragma unroll
            for (int i = 0; i < 16; ++i) bfr[i] = (_Float16)bp[i];
            v8f* accp = tile == 0 ? &acc0 : tile == 1 ? &acc1 : tile == 2 ? &acc2 : &acc3;
            *accp = __builtin_amdgcn_wmma_f32_16x16x32_f16(false, a, false, bfr,
                                                           (short)0, *accp, false, false);
        }
    }
    #pragma unroll
    for (int tile = 0; tile < 4; ++tile) {
        const v8f* accp = tile == 0 ? &acc0 : tile == 1 ? &acc1 : tile == 2 ? &acc2 : &acc3;
        const int d = nb0 + tile * 16 + l15;
        const float bs = bf[d];
        #pragma unroll
        for (int r = 0; r < 8; ++r) {
            int m = m0 + r + half16 * 8;
            x1[(size_t)m * DD + d] = leaky_f((*accp)[r] + bs);
        }
    }
}

// ------------------------------ K3/K4: conv as implicit GEMM (WMMA f16) -----
// Y[b][co][t] = leaky( sum_{ci,dk} W[co][ci*3+dk] * max3(X[b][ci][t+dk..+2]) + bias[co] )
// Block = 4 waves (one co-tile each) x a 32-wide t-slab (2 WMMA t-tiles/wave).
// The raw X slab (Cin x 36 floats) is ASYNC-copied into LDS once per block
// (global_load_async_to_lds_b64, ASYNCcnt), then all B fragments are built
// from LDS with the maxpool3 fused at ds_load time. A (weights) fragment is
// loaded once per k-step and reused by both t-tile WMMAs.
#define TSLAB   32
#define RAWW    36                      // TSLAB + 4 staged floats per ci row
__global__ void conv_wmma_kernel(const float* __restrict__ X, const float* __restrict__ W,
                                 const float* __restrict__ bias, float* __restrict__ Y,
                                 int Cin, int Cout, int LinX, int Lout) {
    __shared__ float raw[256 * RAWW];   // 36,864 B (max Cin = 256)

    const int lane   = threadIdx.x & 31;
    const int wave   = threadIdx.x >> 5;
    const int l15    = lane & 15;
    const int half16 = lane >> 4;
    const int b    = blockIdx.z;
    const int co0  = (blockIdx.y * 4 + wave) * 16;
    const int t0s  = blockIdx.x * TSLAB;
    const int Ktot = Cin * 3;
    const float* Xb = X + (size_t)b * Cin * LinX;

    // --- async-stage raw X slab: raw[ci][j] = X[b][ci][min(t0s+j, safe)] ----
    {
        const unsigned ldsbase = (unsigned)(size_t)&raw[0];
        const int totalPairs = Cin * (RAWW / 2);           // b64 = 2 floats
        for (int p = threadIdx.x; p < totalPairs; p += 128) {
            int ci = p / (RAWW / 2);
            int jp = (p - ci * (RAWW / 2)) * 2;            // even j
            int u0 = t0s + jp;
            if (u0 > LinX - 2) u0 = LinX - 2;              // edge clamp (8B aligned)
            const float* gsrc = Xb + (size_t)ci * LinX + u0;
            unsigned laddr = ldsbase + (unsigned)((ci * RAWW + jp) * 4);
            asm volatile("global_load_async_to_lds_b64 %0, %1, off"
                         :: "v"(laddr), "v"((unsigned long long)(size_t)gsrc)
                         : "memory");
        }
        asm volatile("s_wait_asynccnt 0x0" ::: "memory");
    }
    __syncthreads();

    v8f acc0 = {}, acc1 = {};
    for (int kb = 0; kb < Ktot; kb += 32) {
        // A fragment: weight rows, K contiguous ([co][ci][3] == [co][K])
        v16h a;
        {
            const float* ap = W + (size_t)(co0 + l15) * Ktot + kb + half16 * 16;
            __builtin_prefetch(ap + 32, 0, 1);
            #pragma unroll
            for (int i = 0; i < 8; ++i) a[i]     = (_Float16)ap[i];
            #pragma unroll
            for (int i = 0; i < 8; ++i) a[8 + i] = (_Float16)ap[8 + i];
        }
        const int kbase = kb + half16 * 16;
        #pragma unroll
        for (int tile = 0; tile < 2; ++tile) {
            const int tb = tile * 16 + l15;                // t - t0s for this column
            v16h bfr;
            #pragma unroll
            for (int i = 0; i < 16; ++i) {
                int k  = kbase + i;
                int ci = k / 3;
                int dk = k - ci * 3;
                const float* rp = &raw[ci * RAWW + tb + dk];
                float m  = rp[0];
                float m1 = rp[1]; if (m1 > m) m = m1;
                float m2 = rp[2]; if (m2 > m) m = m2;      // fused maxpool3 (LDS)
                bfr[i] = (_Float16)m;
            }
            if (tile == 0)
                acc0 = __builtin_amdgcn_wmma_f32_16x16x32_f16(false, a, false, bfr,
                                                              (short)0, acc0, false, false);
            else
                acc1 = __builtin_amdgcn_wmma_f32_16x16x32_f16(false, a, false, bfr,
                                                              (short)0, acc1, false, false);
        }
    }
    #pragma unroll
    for (int tile = 0; tile < 2; ++tile) {
        const v8f* accp = tile == 0 ? &acc0 : &acc1;
        const int tt = t0s + tile * 16 + l15;
        if (tt < Lout) {
            #pragma unroll
            for (int r = 0; r < 8; ++r) {
                int co = co0 + r + half16 * 8;
                Y[((size_t)b * Cout + co) * Lout + tt] = leaky_f((*accp)[r] + bias[co]);
            }
        }
    }
}

// ------------------------------ K5: pool + add + LN * c + LN ----------------
__device__ __forceinline__ void block_stats(float psum, float psq,
                                            float* redA, float* redB, float* stats,
                                            int lane, int wave) {
    #pragma unroll
    for (int off = 16; off > 0; off >>= 1) {
        psum += __shfl_down(psum, off, 32);
        psq  += __shfl_down(psq,  off, 32);
    }
    if (lane == 0) { redA[wave] = psum; redB[wave] = psq; }
    __syncthreads();
    if (wave == 0 && lane == 0) {
        float sa = 0.0f, sb = 0.0f;
        #pragma unroll
        for (int i = 0; i < 8; ++i) { sa += redA[i]; sb += redB[i]; }
        float mean = sa * (1.0f / DD);
        float var  = sb * (1.0f / DD) - mean * mean;
        stats[0] = mean;
        stats[1] = rsqrtf(var + 1e-5f);
    }
    __syncthreads();
}

__global__ void finalize_kernel(const float* __restrict__ x1, const float* __restrict__ y3,
                                const float* __restrict__ c,
                                const float* __restrict__ g1, const float* __restrict__ be1,
                                const float* __restrict__ g2, const float* __restrict__ be2,
                                float* __restrict__ out) {
    const int L3 = SEQ + 2;                      // 2038
    const int sb = blockIdx.x;                   // row = s*B + b
    const int b  = sb & (BB - 1);
    const int s  = sb >> 5;
    const int tid = threadIdx.x;                 // 256 threads, 2 d's each
    const int lane = tid & 31, wave = tid >> 5;
    __shared__ float redA[8], redB[8], stats[2];

    float v[2];
    #pragma unroll
    for (int j = 0; j < 2; ++j) {
        int d = tid + j * 256;
        float xv = x1[(size_t)sb * DD + d];
        const float* yp = y3 + ((size_t)b * DD + d) * L3 + s;
        float m  = yp[0];
        float m1 = yp[1]; if (m1 > m) m = m1;
        float m2 = yp[2]; if (m2 > m) m = m2;    // maxpool3 over seq, transposed read
        v[j] = xv + m;
    }
    block_stats(v[0] + v[1], v[0] * v[0] + v[1] * v[1], redA, redB, stats, lane, wave);
    float mean = stats[0], inv = stats[1];
    #pragma unroll
    for (int j = 0; j < 2; ++j) {
        int d = tid + j * 256;
        v[j] = ((v[j] - mean) * inv * g1[d] + be1[d]) * c[b * DD + d];
    }
    __syncthreads();
    block_stats(v[0] + v[1], v[0] * v[0] + v[1] * v[1], redA, redB, stats, lane, wave);
    mean = stats[0]; inv = stats[1];
    #pragma unroll
    for (int j = 0; j < 2; ++j) {
        int d = tid + j * 256;
        out[(size_t)sb * DD + d] = (v[j] - mean) * inv * g2[d] + be2[d];
    }
}

// ------------------------------ launcher ------------------------------------
extern "C" void kernel_launch(void* const* d_in, const int* in_sizes, int n_in,
                              void* d_out, int out_size, void* d_ws, size_t ws_size,
                              hipStream_t stream) {
    const float* input   = (const float*)d_in[0];   // [2036,32,64]
    const float* history = (const float*)d_in[1];   // [32,4,2048]
    const float* statc   = (const float*)d_in[2];   // [32,100]
    const float* W_feat  = (const float*)d_in[3];   // [512,64]
    const float* b_feat  = (const float*)d_in[4];   // [512]
    const float* w1      = (const float*)d_in[5];   // [128,4,3]
    const float* b1      = (const float*)d_in[6];   // [128]
    const float* w2      = (const float*)d_in[7];   // [256,128,3]
    const float* b2      = (const float*)d_in[8];   // [256]
    const float* w3      = (const float*)d_in[9];   // [512,256,3]
    const float* b3      = (const float*)d_in[10];  // [512]
    const float* g1      = (const float*)d_in[11];
    const float* beta1   = (const float*)d_in[12];
    const float* W_city  = (const float*)d_in[13];  // [512,100]
    const float* b_city  = (const float*)d_in[14];
    const float* g2      = (const float*)d_in[15];
    const float* beta2   = (const float*)d_in[16];
    float* out = (float*)d_out;

    // Workspace layout (floats). y1 aliases the y3 slab (y1 dies before conv3 writes y3).
    float* ws = (float*)d_ws;
    const size_t M = (size_t)SEQ * BB;                   // 65152
    float* c_ws = ws;                                    // 32*512
    float* x1   = c_ws + (size_t)BB * DD;                // 65152*512
    float* y3   = x1 + M * DD;                           // 32*512*2038
    float* y1   = y3;                                    // 32*128*2046 (aliased)
    float* y2   = y3 + (size_t)BB * DD * (SEQ + 2);      // 32*256*2042

    // K0: city embed
    city_kernel<<<dim3((BB * DD + 255) / 256), dim3(256), 0, stream>>>(
        statc, W_city, b_city, c_ws);

    // K1: conv1 (scalar, tiny K)
    {
        size_t total = (size_t)BB * 128 * (LL - 2);
        conv1_kernel<<<dim3((unsigned)((total + 255) / 256)), dim3(256), 0, stream>>>(
            history, w1, b1, y1);
    }

    // K2: feature GEMM: grid (1018 m-blocks, 8 n-groups), 4 waves/block, 8 WMMA/wave
    feat_gemm_kernel<<<dim3(1018, 8), dim3(128), 0, stream>>>(input, W_feat, b_feat, x1);

    // K3: conv2: Cin=128, Cout=256, LinX=2046, Lout=2042; 64 t-slabs of 32
    conv_wmma_kernel<<<dim3(64, 4, BB), dim3(128), 0, stream>>>(
        y1, w2, b2, y2, 128, 256, LL - 2, LL - 6);

    // K4: conv3: Cin=256, Cout=512, LinX=2042, Lout=2038
    conv_wmma_kernel<<<dim3(64, 8, BB), dim3(128), 0, stream>>>(
        y2, w3, b3, y3, 256, 512, LL - 6, LL - 10);

    // K5: pool + add + LN * c + LN -> out
    finalize_kernel<<<dim3((unsigned)M), dim3(256), 0, stream>>>(
        x1, y3, c_ws, g1, beta1, g2, beta2, out);

    (void)in_sizes; (void)n_in; (void)out_size; (void)ws_size;
}
